// GRUDLayer_66082366816552
// MI455X (gfx1250) — compile-verified, matrix operand order checked by reading, not run on hardware
//
#include <hip/hip_runtime.h>
#include <hip/hip_bf16.h>

// GRU-D persistent-recurrence kernel for gfx1250 (MI455X).
// - 32 blocks x 16 batch rows; 16 waves/block; each wave owns a 16x32 slice of
//   h in v_wmma f32 accumulators across all T=256 steps (h never leaves VGPRs).
// - All GEMMs via V_WMMA_F32_16X16X32_F16; weights pre-converted to f16
//   (2.4 MB, L2-resident at 192 MB L2).
// - Per-step input tiles staged into LDS by the Tensor Data Mover
//   (tensor_load_to_lds + s_wait_tensorcnt), overlapped with the GEMM stages.
// - Transcendentals via v_exp_f32 / v_rcp_f32 instead of libm chains.

typedef _Float16 half8   __attribute__((ext_vector_type(8)));
typedef _Float16 half16  __attribute__((ext_vector_type(16)));
typedef float    float8  __attribute__((ext_vector_type(8)));
typedef unsigned int u32x4 __attribute__((ext_vector_type(4)));
typedef int      i32x4  __attribute__((ext_vector_type(4)));
typedef int      i32x8  __attribute__((ext_vector_type(8)));

#define Bsz   512
#define Tsz   256
#define Dsz   128
#define Hsz   512
#define GIN   768   // 2*D + H
#define GINP  776   // LDS row stride (halfs): +16B pad -> conflict-free ds_load_b128
#define DPAD  136   // padded D row stride for delta tile
#define BM    16    // batch rows per block (one WMMA M-tile)
#define NTHREADS 512  // 16 waves; each wave covers 32 H-columns (2 N-tiles)

#if defined(__has_builtin)
#  if __has_builtin(__builtin_amdgcn_tensor_load_to_lds) && __has_builtin(__builtin_amdgcn_s_wait_tensorcnt)
#    define USE_TDM 1
#  else
#    define USE_TDM 0
#  endif
#else
#  define USE_TDM 0
#endif

// ---------------- fast math (hardware trans ops) ----------------
__device__ __forceinline__ float fast_rcp(float x) {
#if defined(__has_builtin) && __has_builtin(__builtin_amdgcn_rcpf)
    return __builtin_amdgcn_rcpf(x);
#else
    return 1.0f / x;
#endif
}
__device__ __forceinline__ float fast_sigmoid(float v) {
    return fast_rcp(1.0f + __expf(-v));                 // v_exp_f32 + v_rcp_f32
}
__device__ __forceinline__ float fast_tanh(float v) {
    const float e = __expf(2.0f * v);                   // inf/0 saturate to +-1
    return 1.0f - 2.0f * fast_rcp(e + 1.0f);
}

// Build a 16-half A/B fragment per ISA 7.12.2 (16-bit 16x32 layout):
//   lanes 0-15 : row = lane,     K = k..k+7   and k+16..k+23
//   lanes 16-31: row = lane-16,  K = k+8..k+15 and k+24..k+31
__device__ __forceinline__ half16 load_frag(const _Float16* base, int rowStride,
                                            int kbase, int lane) {
    const int lm  = lane & 15;
    const int sel = (lane >> 4) & 1;
    const _Float16* p = base + (size_t)lm * rowStride + kbase + sel * 8;
    half8 lo = *(const half8*)(p);
    half8 hi = *(const half8*)(p + 16);
    return __builtin_shufflevector(lo, hi, 0,1,2,3,4,5,6,7,8,9,10,11,12,13,14,15);
}

__device__ __forceinline__ float8 wmma16(half16 a, half16 b, float8 c) {
    return __builtin_amdgcn_wmma_f32_16x16x32_f16(false, a, false, b, (short)0, c,
                                                  false, false);
}

#if USE_TDM
// TDM: DMA one (BM x Dsz) f32 tile (row stride T*D elements) from global to LDS.
// D# per ISA 8.3/8.4: group0 = {count/type/addrs}, group1 = {dims/tile/strides}.
__device__ __forceinline__ void tdm_load_tile_f32(const float* gsrc, unsigned lds_off) {
    const unsigned long long ga = (unsigned long long)(uintptr_t)gsrc;
    u32x4 g0;
    g0[0] = 1u;                                          // count=1 user descriptor
    g0[1] = lds_off;                                     // lds_addr (bytes)
    g0[2] = (unsigned)(ga & 0xFFFFFFFFu);                // global_addr[31:0]
    g0[3] = (unsigned)((ga >> 32) & 0x01FFFFFFu)         // global_addr[56:32]
          | (2u << 30);                                  // type = 2 ("image")
    i32x8 g1;
    g1[0] = (2 << 16);                                   // data_size = 2 -> 4 bytes
    g1[1] = (Dsz & 0xFFFF) << 16;                        // tensor_dim0[15:0]
    g1[2] = (BM & 0xFFFF) << 16;                         // tensor_dim0 hi=0 | tensor_dim1 lo
    g1[3] = (Dsz & 0xFFFF) << 16;                        // tensor_dim1 hi=0 | tile_dim0=128
    g1[4] = BM;                                          // tile_dim1=16 | tile_dim2=0
    g1[5] = Tsz * Dsz;                                   // tensor_dim0_stride (elements)
    g1[6] = 0;
    g1[7] = 0;
    const i32x4 z4 = {0, 0, 0, 0};
#if __clang_major__ >= 23
    const i32x8 z8 = {0, 0, 0, 0, 0, 0, 0, 0};
    __builtin_amdgcn_tensor_load_to_lds(g0, g1, z4, z4, z8, 0);
#else
    __builtin_amdgcn_tensor_load_to_lds(g0, g1, z4, z4, 0);
#endif
}
#endif  // USE_TDM

__global__ void convert_weights(const float* __restrict__ Wz, const float* __restrict__ Wr,
                                const float* __restrict__ Wn, const float* __restrict__ Wgh,
                                const float* __restrict__ Wgx,
                                _Float16* __restrict__ Wz16, _Float16* __restrict__ Wr16,
                                _Float16* __restrict__ Wn16, _Float16* __restrict__ Wgh16,
                                float* __restrict__ wgxd) {
    const int i = blockIdx.x * blockDim.x + threadIdx.x;
    if (i < Hsz * GIN) {
        Wz16[i] = (_Float16)Wz[i];
        Wr16[i] = (_Float16)Wr[i];
        Wn16[i] = (_Float16)Wn[i];
    }
    if (i < Hsz * Dsz) Wgh16[i] = (_Float16)Wgh[i];
    if (i < Dsz)       wgxd[i]  = Wgx[(size_t)i * Dsz + i];   // diagonal of Wgx
}

__global__ __launch_bounds__(NTHREADS, 1)
void grud_recurrence(const float* __restrict__ x,   const float* __restrict__ xl,
                     const float* __restrict__ msk, const float* __restrict__ dlt,
                     const float* __restrict__ xmean,
                     const float* __restrict__ bz,  const float* __restrict__ br,
                     const float* __restrict__ bn,  const float* __restrict__ bgx,
                     const float* __restrict__ bgh,
                     const _Float16* __restrict__ Wz16, const _Float16* __restrict__ Wr16,
                     const _Float16* __restrict__ Wn16, const _Float16* __restrict__ Wgh16,
                     const float* __restrict__ wgxd,
                     float* __restrict__ out) {
    __shared__ __align__(16) _Float16 sComb [BM * GINP];  // [x_tilde | h   | m]
    __shared__ __align__(16) _Float16 sCombN[BM * GINP];  // [x_tilde | r*h | m]
    __shared__ __align__(16) _Float16 sDt   [BM * DPAD];  // delta (f16) for gh GEMM
#if USE_TDM
    __shared__ __align__(16) float    sIn[4][BM * Dsz];   // TDM-staged x, xl, m, delta
#endif

    const int tid   = threadIdx.x;
    const int lane  = tid & 31;
    const int wave  = tid >> 5;
    const int lm    = lane & 15;
    const int rbase = (lane >> 4) * 8;         // C/D frag: row = j + 8*(lane>=16)
    const int b0    = blockIdx.x * BM;
    const int n0    = wave * 32;               // this wave's H-column base

    const float8 f8z = {0.f, 0.f, 0.f, 0.f, 0.f, 0.f, 0.f, 0.f};
    float8 h[2] = {f8z, f8z};                  // h lives in accumulators all T steps

    // Stage-1 per-thread channel is invariant: d = tid & 127 for every strided
    // iteration (stride 512 == 0 mod 128) -> hoist all per-channel constants.
    const int   dfix  = tid & (Dsz - 1);
    const int   row0  = tid >> 7;              // rows row0 + {0,4,8,12}
    const float wgx_d = wgxd[dfix];
    const float bgx_d = bgx[dfix];
    const float xm_d  = xmean[dfix];

    float bz_c[2], br_c[2], bn_c[2], bgh_c[2];
#pragma unroll
    for (int t0 = 0; t0 < 2; ++t0) {
        const int col = n0 + 16 * t0 + lm;
        bz_c[t0] = bz[col]; br_c[t0] = br[col];
        bn_c[t0] = bn[col]; bgh_c[t0] = bgh[col];
    }

#if USE_TDM
    // Prime the pipeline: DMA step-0 tiles into LDS.
    if (wave == 0) {
        const size_t tile0 = (size_t)b0 * Tsz * Dsz;     // + t*Dsz, t = 0
        tdm_load_tile_f32(x   + tile0, (unsigned)(uintptr_t)(void*)&sIn[0][0]);
        tdm_load_tile_f32(xl  + tile0, (unsigned)(uintptr_t)(void*)&sIn[1][0]);
        tdm_load_tile_f32(msk + tile0, (unsigned)(uintptr_t)(void*)&sIn[2][0]);
        tdm_load_tile_f32(dlt + tile0, (unsigned)(uintptr_t)(void*)&sIn[3][0]);
    }
#endif

    for (int t = 0; t < Tsz; ++t) {
#if USE_TDM
        if (wave == 0) __builtin_amdgcn_s_wait_tensorcnt(0);  // step-t tiles landed
#endif
        __syncthreads();  // TDM visible to all waves; prior n-GEMM reads done

        // ---- Stage 1: elementwise input decay, stage comb tiles in LDS ----
#pragma unroll
        for (int rr = 0; rr < 4; ++rr) {
            const int row = row0 + rr * 4;
            const int i   = row * Dsz + dfix;
#if USE_TDM
            const float xv = sIn[0][i], xlv = sIn[1][i];
            const float mv = sIn[2][i], dv  = sIn[3][i];
#else
            const size_t g = ((size_t)(b0 + row) * Tsz + t) * Dsz + dfix;
            const float xv = x[g], xlv = xl[g], mv = msk[g], dv = dlt[g];
#endif
            const float gx = __expf(-fmaxf(dv * wgx_d + bgx_d, 0.0f));
            const float xt = mv * xv + (1.0f - mv) * (gx * xlv + (1.0f - gx) * xm_d);
            const _Float16 xth = (_Float16)xt;
            const _Float16 mh  = (_Float16)mv;
            sComb [row * GINP + dfix] = xth;
            sCombN[row * GINP + dfix] = xth;
            sComb [row * GINP + Dsz + Hsz + dfix] = mh;
            sCombN[row * GINP + Dsz + Hsz + dfix] = mh;
            sDt[row * DPAD + dfix] = (_Float16)dv;
        }
        __syncthreads();

#if USE_TDM
        // Kick off step t+1 DMA now; it overlaps the three GEMM stages below.
        if (wave == 0 && (t + 1) < Tsz) {
            const size_t tile = ((size_t)b0 * Tsz + (t + 1)) * Dsz;
            tdm_load_tile_f32(x   + tile, (unsigned)(uintptr_t)(void*)&sIn[0][0]);
            tdm_load_tile_f32(xl  + tile, (unsigned)(uintptr_t)(void*)&sIn[1][0]);
            tdm_load_tile_f32(msk + tile, (unsigned)(uintptr_t)(void*)&sIn[2][0]);
            tdm_load_tile_f32(dlt + tile, (unsigned)(uintptr_t)(void*)&sIn[3][0]);
        }
#endif

        // ---- Stage 2: gamma_h = exp(-relu(dt @ Wgh.T + bgh)); h *= gamma_h ----
#pragma unroll
        for (int t0 = 0; t0 < 2; ++t0) {
            float8 acc = f8z;
            const _Float16* wb = Wgh16 + (size_t)(n0 + 16 * t0) * Dsz;
#pragma unroll
            for (int k = 0; k < Dsz; k += 32) {
                half16 a = load_frag(sDt, DPAD, k, lane);
                half16 b = load_frag(wb, Dsz, k, lane);
                acc = wmma16(a, b, acc);
            }
            const int col = Dsz + n0 + 16 * t0 + lm;
#pragma unroll
            for (int j = 0; j < 8; ++j) {
                const float gh = __expf(-fmaxf(acc[j] + bgh_c[t0], 0.0f));
                const float hv = h[t0][j] * gh;
                h[t0][j] = hv;
                sComb[(rbase + j) * GINP + col] = (_Float16)hv;  // h -> comb (f16)
            }
        }
        __syncthreads();

        // ---- Stage 3: z, r = sigmoid(comb @ {Wz,Wr}.T + b); stage r*h ----
        float8 zacc[2] = {f8z, f8z};
        float8 racc[2] = {f8z, f8z};
#pragma unroll 4
        for (int k = 0; k < GIN; k += 32) {
            half16 a = load_frag(sComb, GINP, k, lane);
#pragma unroll
            for (int t0 = 0; t0 < 2; ++t0) {
                const size_t wro = (size_t)(n0 + 16 * t0) * GIN;
                half16 bzf = load_frag(Wz16 + wro, GIN, k, lane);
                zacc[t0] = wmma16(a, bzf, zacc[t0]);
                half16 brf = load_frag(Wr16 + wro, GIN, k, lane);
                racc[t0] = wmma16(a, brf, racc[t0]);
            }
        }
        float8 z[2];
#pragma unroll
        for (int t0 = 0; t0 < 2; ++t0) {
            const int col = Dsz + n0 + 16 * t0 + lm;
#pragma unroll
            for (int j = 0; j < 8; ++j) {
                const float zv = fast_sigmoid(zacc[t0][j] + bz_c[t0]);
                const float rv = fast_sigmoid(racc[t0][j] + br_c[t0]);
                z[t0][j] = zv;
                sCombN[(rbase + j) * GINP + col] = (_Float16)(rv * h[t0][j]);
            }
        }
        __syncthreads();

        // ---- Stage 4: h_tilde = tanh(comb_r @ Wn.T + bn); h = (1-z)h + z*h_t ----
        float8 nacc[2] = {f8z, f8z};
#pragma unroll 4
        for (int k = 0; k < GIN; k += 32) {
            half16 a = load_frag(sCombN, GINP, k, lane);
#pragma unroll
            for (int t0 = 0; t0 < 2; ++t0) {
                half16 bnf = load_frag(Wn16 + (size_t)(n0 + 16 * t0) * GIN, GIN, k, lane);
                nacc[t0] = wmma16(a, bnf, nacc[t0]);
            }
        }
#pragma unroll
        for (int t0 = 0; t0 < 2; ++t0) {
#pragma unroll
            for (int j = 0; j < 8; ++j) {
                const float ht = fast_tanh(nacc[t0][j] + bn_c[t0]);
                const float zv = z[t0][j];
                h[t0][j] = (1.0f - zv) * h[t0][j] + zv * ht;
            }
        }
    }

    // ---- Write h_final (B, H), f32 ----
#pragma unroll
    for (int t0 = 0; t0 < 2; ++t0) {
        const int col = n0 + 16 * t0 + lm;
#pragma unroll
        for (int j = 0; j < 8; ++j) {
            out[(size_t)(b0 + rbase + j) * Hsz + col] = h[t0][j];
        }
    }
}

extern "C" void kernel_launch(void* const* d_in, const int* in_sizes, int n_in,
                              void* d_out, int out_size, void* d_ws, size_t ws_size,
                              hipStream_t stream) {
    const float* x     = (const float*)d_in[0];
    const float* xl    = (const float*)d_in[1];
    const float* msk   = (const float*)d_in[2];
    const float* dlt   = (const float*)d_in[3];
    const float* xmean = (const float*)d_in[4];
    const float* Wz    = (const float*)d_in[5];
    const float* bz    = (const float*)d_in[6];
    const float* Wr    = (const float*)d_in[7];
    const float* br    = (const float*)d_in[8];
    const float* Wn    = (const float*)d_in[9];
    const float* bn    = (const float*)d_in[10];
    const float* Wgx   = (const float*)d_in[11];
    const float* bgx   = (const float*)d_in[12];
    const float* Wgh   = (const float*)d_in[13];
    const float* bgh   = (const float*)d_in[14];

    // Workspace: f16 weights (L2-resident, ~2.4 MB) + diag(Wgx) as f32.
    _Float16* Wz16  = (_Float16*)d_ws;
    _Float16* Wr16  = Wz16 + (size_t)Hsz * GIN;
    _Float16* Wn16  = Wr16 + (size_t)Hsz * GIN;
    _Float16* Wgh16 = Wn16 + (size_t)Hsz * GIN;
    float*    wgxd  = (float*)(Wgh16 + (size_t)Hsz * Dsz);

    const int nconv = Hsz * GIN;
    convert_weights<<<(nconv + 255) / 256, 256, 0, stream>>>(
        Wz, Wr, Wn, Wgh, Wgx, Wz16, Wr16, Wn16, Wgh16, wgxd);

    grud_recurrence<<<Bsz / BM, NTHREADS, 0, stream>>>(
        x, xl, msk, dlt, xmean, bz, br, bn, bgx, bgh,
        Wz16, Wr16, Wn16, Wgh16, wgxd, (float*)d_out);
}
